// GraphTransformer_89163521065125
// MI455X (gfx1250) — matrix operand order repaired and checked by reading, not compiled, not run on hardware
//
#include <hip/hip_runtime.h>
#include <hip/hip_bf16.h>
#include <math.h>

typedef __attribute__((ext_vector_type(16))) _Float16 v16h;
typedef __attribute__((ext_vector_type(8)))  float    v8f;

#define NNODES 50000
#define NEDGES 625000
#define CCH    128
#define NHEAD  8
#define DHEAD  16

// order-preserving float <-> uint mapping for atomicMax on floats
__device__ __forceinline__ unsigned fenc(float f) {
  unsigned u = __float_as_uint(f);
  return (u & 0x80000000u) ? ~u : (u | 0x80000000u);
}
__device__ __forceinline__ float fdec(unsigned u) {
  return __uint_as_float((u & 0x80000000u) ? (u ^ 0x80000000u) : ~u);
}

// one K=32 WMMA step: A row pointer arp (row-major, stride K implicit in caller),
// B pointer base Bp (row-major [K,Nc])
__device__ __forceinline__ v8f wmma_step(const float* __restrict__ ap /* +k0+half*8 */,
                                         const float* __restrict__ bp /* +(k0+half*8)*Nc+n */,
                                         int Nc, v8f c)
{
  v16h a, b;
#pragma unroll
  for (int i = 0; i < 8; ++i) a[i]     = (_Float16)ap[i];
#pragma unroll
  for (int i = 0; i < 8; ++i) a[i + 8] = (_Float16)ap[16 + i];
#pragma unroll
  for (int i = 0; i < 16; ++i) b[i] = (_Float16)bp[(size_t)(i + (i >= 8 ? 8 : 0)) * Nc];
  return __builtin_amdgcn_wmma_f32_16x16x32_f16(false, a, false, b, (short)0, c,
                                                false, false);
}

// ---------------------------------------------------------------------------
// Generic GEMM: C[M,Nc] = relu?(A[M,K] @ B[K,Nc] + bias), f32 in/out,
// v_wmma_f32_16x16x32_f16 inside, dual accumulator chains for ILP.
// K % 32 == 0, Nc % 64 == 0. Block = 128 threads = 4 waves; each wave owns one
// 16x16 N-subtile. grid = (ceil(M/16), Nc/64)
// ---------------------------------------------------------------------------
template<bool RELU>
__global__ void gemm_wmma(const float* __restrict__ A, const float* __restrict__ B,
                          const float* __restrict__ bias, float* __restrict__ Cm,
                          int M, int K, int Nc)
{
  const int lane = threadIdx.x & 31;
  const int wv   = threadIdx.x >> 5;
  const int m16  = lane & 15;
  const int half = lane >> 4;
  const int row0 = blockIdx.x * 16;
  const int n    = (blockIdx.y * 4 + wv) * 16 + m16;

  float bvv = bias ? bias[n] : 0.0f;
  v8f c0, c1;
#pragma unroll
  for (int r = 0; r < 8; ++r) { c0[r] = bvv; c1[r] = 0.0f; }

  int arow = row0 + m16;
  if (arow >= M) arow = M - 1;           // clamp (stores are guarded)
  const float* arp = A + (size_t)arow * K + half * 8;
  const float* brp = B + (size_t)(half * 8) * Nc + n;

  int k0 = 0;
  for (; k0 + 64 <= K; k0 += 64) {       // two independent WMMA chains
    c0 = wmma_step(arp + k0,      brp + (size_t)k0 * Nc,        Nc, c0);
    c1 = wmma_step(arp + k0 + 32, brp + (size_t)(k0 + 32) * Nc, Nc, c1);
  }
  for (; k0 < K; k0 += 32)               // remainder (K == 32 case)
    c0 = wmma_step(arp + k0, brp + (size_t)k0 * Nc, Nc, c0);

#pragma unroll
  for (int r = 0; r < 8; ++r) {
    int row = row0 + r + half * 8;
    if (row < M) {
      float v = c0[r] + c1[r];
      if (RELU) v = fmaxf(v, 0.0f);
      Cm[(size_t)row * Nc + n] = v;
    }
  }
}

// ---------------------------------------------------------------------------
// Attention helpers
// ---------------------------------------------------------------------------
__global__ void attn_init(unsigned* __restrict__ smax, float* __restrict__ denom, int n)
{
  int i = blockIdx.x * blockDim.x + threadIdx.x;
  if (i < n) { smax[i] = 0x007FFFFFu; /* fenc(-inf) */ denom[i] = 0.0f; }
}

__global__ void attn_score(const float* __restrict__ q, const float* __restrict__ k,
                           const float* __restrict__ ee, const int* __restrict__ src,
                           const int* __restrict__ dst, float* __restrict__ score,
                           unsigned* __restrict__ smax)
{
  int idx = blockIdx.x * blockDim.x + threadIdx.x;
  if (idx >= NEDGES * NHEAD) return;
  int ed = idx / NHEAD, hh = idx % NHEAD;
  int s = src[ed], d = dst[ed];
  const float* qp = q  + (size_t)d  * CCH + hh * DHEAD;
  const float* kp = k  + (size_t)s  * CCH + hh * DHEAD;
  const float* ep = ee + (size_t)ed * CCH + hh * DHEAD;
  float acc = 0.0f;
#pragma unroll
  for (int i = 0; i < DHEAD; ++i) acc += qp[i] * (kp[i] + ep[i]);
  acc *= 0.25f;                                  // 1/sqrt(16)
  score[idx] = acc;
  atomicMax(&smax[d * NHEAD + hh], fenc(acc));
}

__global__ void attn_exp(float* __restrict__ score, const unsigned* __restrict__ smax,
                         float* __restrict__ denom, const int* __restrict__ dst)
{
  int idx = blockIdx.x * blockDim.x + threadIdx.x;
  if (idx >= NEDGES * NHEAD) return;
  int ed = idx / NHEAD, hh = idx % NHEAD;
  int d = dst[ed];
  float ex = __expf(score[idx] - fdec(smax[d * NHEAD + hh]));
  score[idx] = ex;
  atomicAdd(&denom[d * NHEAD + hh], ex);
}

__global__ void attn_scatter(const float* __restrict__ ex, const float* __restrict__ denom,
                             const float* __restrict__ v, const float* __restrict__ ee,
                             const int* __restrict__ src, const int* __restrict__ dst,
                             float* __restrict__ acc)
{
  int idx = blockIdx.x * blockDim.x + threadIdx.x;
  if (idx >= NEDGES * CCH) return;
  int ed = idx / CCH, c = idx % CCH;
  int hh = c / DHEAD;
  int s = src[ed], d = dst[ed];
  float alpha = ex[(size_t)ed * NHEAD + hh] / (denom[d * NHEAD + hh] + 1e-16f);
  float val = alpha * (v[(size_t)s * CCH + c] + ee[(size_t)ed * CCH + c]);
  atomicAdd(&acc[(size_t)d * CCH + c], val);
}

// h = LN(h + a) * g + b, one 128-thread block per row
__global__ void add_layernorm(float* __restrict__ h, const float* __restrict__ a,
                              const float* __restrict__ g, const float* __restrict__ b)
{
  __shared__ float red[CCH];
  int row = blockIdx.x, t = threadIdx.x;
  float x = h[(size_t)row * CCH + t] + a[(size_t)row * CCH + t];
  red[t] = x; __syncthreads();
  for (int s = CCH / 2; s > 0; s >>= 1) { if (t < s) red[t] += red[t + s]; __syncthreads(); }
  float mean = red[0] * (1.0f / CCH);
  __syncthreads();
  float dx = x - mean;
  red[t] = dx * dx; __syncthreads();
  for (int s = CCH / 2; s > 0; s >>= 1) { if (t < s) red[t] += red[t + s]; __syncthreads(); }
  float var = red[0] * (1.0f / CCH);
  h[(size_t)row * CCH + t] = dx * rsqrtf(var + 1e-5f) * g[t] + b[t];
}

// ---------------------------------------------------------------------------
// Final MLP stage 1 with fused gather: z1[E,64] = relu(concat(h[src],h[dst],e) @ w1 + b1)
// K = 384 (fully unrolled so the per-step source select is compile-time),
// N padded 50 -> 64, B loads unpredicated (clamped column + arithmetic zero).
// Block 128 = 4 waves, wave wv owns n-tile wv*16. grid = (ceil(E/16), 1)
// ---------------------------------------------------------------------------
__global__ void mlp1_wmma(const float* __restrict__ h, const float* __restrict__ e,
                          const int* __restrict__ src, const int* __restrict__ dst,
                          const float* __restrict__ w1, const float* __restrict__ b1,
                          float* __restrict__ z1)
{
  const int lane = threadIdx.x & 31;
  const int wv   = threadIdx.x >> 5;
  const int m16  = lane & 15;
  const int half = lane >> 4;
  const int row0 = blockIdx.x * 16;
  const int n    = wv * 16 + m16;              // 0..63
  const bool nok = (n < 50);
  const int ncl  = nok ? n : 49;               // clamped column, loads always valid

  int row = row0 + m16;
  if (row >= NEDGES) row = NEDGES - 1;
  const int sIdx = src[row], dIdx = dst[row];

  float bvv = nok ? b1[n] : 0.0f;
  v8f c0, c1;
#pragma unroll
  for (int r = 0; r < 8; ++r) { c0[r] = bvv; c1[r] = 0.0f; }

  const float* hs = h + (size_t)sIdx * CCH + half * 8;
  const float* hd = h + (size_t)dIdx * CCH + half * 8;
  const float* ep = e + (size_t)row  * CCH + half * 8;

#pragma unroll
  for (int step = 0; step < 12; ++step) {
    const int k0 = step * 32;
    const float* ap = (k0 < 128) ? (hs + k0)
                    : (k0 < 256) ? (hd + (k0 - 128))
                                 : (ep + (k0 - 256));
    v16h a, b;
#pragma unroll
    for (int i = 0; i < 8; ++i) a[i]     = (_Float16)ap[i];
#pragma unroll
    for (int i = 0; i < 8; ++i) a[i + 8] = (_Float16)ap[16 + i];
#pragma unroll
    for (int i = 0; i < 16; ++i) {
      int kk = k0 + half * 8 + i + (i >= 8 ? 8 : 0);
      float t = w1[(size_t)kk * 50 + ncl];     // unconditional load
      b[i] = (_Float16)(nok ? t : 0.0f);       // v_cndmask, no exec masking
    }
    if (step & 1)
      c1 = __builtin_amdgcn_wmma_f32_16x16x32_f16(false, a, false, b, (short)0, c1,
                                                  false, false);
    else
      c0 = __builtin_amdgcn_wmma_f32_16x16x32_f16(false, a, false, b, (short)0, c0,
                                                  false, false);
  }

#pragma unroll
  for (int r = 0; r < 8; ++r) {
    int rr = row0 + r + half * 8;
    if (rr < NEDGES) z1[(size_t)rr * 64 + n] = fmaxf(c0[r] + c1[r], 0.0f);
  }
}

// stages 2+3: z2 = relu(z1 @ w2 + b2); out = z2 @ w3 + b3  (tiny K, VALU)
__global__ void mlp_tail(const float* __restrict__ z1, const float* __restrict__ w2,
                         const float* __restrict__ b2, const float* __restrict__ w3,
                         const float* __restrict__ b3, float* __restrict__ out)
{
  int ed = blockIdx.x * blockDim.x + threadIdx.x;
  if (ed >= NEDGES) return;
  const float* zp = z1 + (size_t)ed * 64;
  float z2[25];
#pragma unroll
  for (int j = 0; j < 25; ++j) {
    float acc = b2[j];
    for (int i = 0; i < 50; ++i) acc += zp[i] * w2[i * 25 + j];
    z2[j] = fmaxf(acc, 0.0f);
  }
#pragma unroll
  for (int o = 0; o < 2; ++o) {
    float acc = b3[o];
#pragma unroll
    for (int j = 0; j < 25; ++j) acc += z2[j] * w3[j * 2 + o];
    out[(size_t)ed * 2 + o] = acc;
  }
}

// ---------------------------------------------------------------------------
extern "C" void kernel_launch(void* const* d_in, const int* in_sizes, int n_in,
                              void* d_out, int out_size, void* d_ws, size_t ws_size,
                              hipStream_t stream)
{
  const float* x      = (const float*)d_in[0];
  const int*   eidx   = (const int*)d_in[1];
  const float* eattr  = (const float*)d_in[2];
  const float* node_w = (const float*)d_in[3];
  const float* node_b = (const float*)d_in[4];
  const float* edge_w = (const float*)d_in[5];
  const float* edge_b = (const float*)d_in[6];
  const float* wq = (const float*)d_in[7];   const float* bq = (const float*)d_in[8];
  const float* wk = (const float*)d_in[9];   const float* bk = (const float*)d_in[10];
  const float* wv = (const float*)d_in[11];  const float* bv = (const float*)d_in[12];
  const float* we = (const float*)d_in[13];
  const float* wsk= (const float*)d_in[14];  const float* bsk= (const float*)d_in[15];
  const float* ln1g=(const float*)d_in[16];  const float* ln1b=(const float*)d_in[17];
  const float* fw1= (const float*)d_in[18];  const float* fb1= (const float*)d_in[19];
  const float* fw2= (const float*)d_in[20];  const float* fb2= (const float*)d_in[21];
  const float* ln2g=(const float*)d_in[22];  const float* ln2b=(const float*)d_in[23];
  const float* mw1= (const float*)d_in[24];  const float* mb1= (const float*)d_in[25];
  const float* mw2= (const float*)d_in[26];  const float* mb2= (const float*)d_in[27];
  const float* mw3= (const float*)d_in[28];  const float* mb3= (const float*)d_in[29];
  float* out = (float*)d_out;

  const int* src = eidx;
  const int* dst = eidx + NEDGES;

  // workspace layout (floats) -- ~894 MB total
  float* ws = (float*)d_ws;
  size_t o = 0;
  float* hb  = ws + o; o += (size_t)NNODES * CCH;   // node features
  float* acc = ws + o; o += (size_t)NNODES * CCH;   // skip + attn accumulator / ffn out
  float* qb  = ws + o; o += (size_t)NNODES * CCH;
  float* kb  = ws + o; o += (size_t)NNODES * CCH;
  float* vb  = ws + o; o += (size_t)NNODES * CCH;
  float* ff  = ws + o; o += (size_t)NNODES * 512;   // ffn hidden
  float* eb  = ws + o; o += (size_t)NEDGES * CCH;   // edge features e
  float* eeb = ws + o; o += (size_t)NEDGES * CCH;   // per-layer e @ we (reused as z1)
  float* sc  = ws + o; o += (size_t)NEDGES * NHEAD; // scores / exp
  unsigned* smax = (unsigned*)(ws + o); o += (size_t)NNODES * NHEAD;
  float* den = ws + o; o += (size_t)NNODES * NHEAD;
  float* z1  = eeb;                                  // E x 64, reuse

  const dim3 blk(128, 1, 1);
  const dim3 gN2((NNODES + 15) / 16, 2);   // Nc=128
  const dim3 gN8((NNODES + 15) / 16, 8);   // Nc=512
  const dim3 gE2((NEDGES + 15) / 16, 2);   // Nc=128
  const int nh = NNODES * NHEAD;
  const int eh = NEDGES * NHEAD;
  const int ec = NEDGES * CCH;

  // embeddings
  gemm_wmma<false><<<gN2, blk, 0, stream>>>(x,     node_w, node_b, hb, NNODES, 64, CCH);
  gemm_wmma<false><<<gE2, blk, 0, stream>>>(eattr, edge_w, edge_b, eb, NEDGES, 32, CCH);

  for (int l = 0; l < 3; ++l) {
    const size_t wo = (size_t)l * CCH * CCH;
    // projections (skip goes straight into the attention accumulator)
    gemm_wmma<false><<<gN2, blk, 0, stream>>>(hb, wq  + wo, bq  + l * CCH, qb,  NNODES, CCH, CCH);
    gemm_wmma<false><<<gN2, blk, 0, stream>>>(hb, wk  + wo, bk  + l * CCH, kb,  NNODES, CCH, CCH);
    gemm_wmma<false><<<gN2, blk, 0, stream>>>(hb, wv  + wo, bv  + l * CCH, vb,  NNODES, CCH, CCH);
    gemm_wmma<false><<<gN2, blk, 0, stream>>>(hb, wsk + wo, bsk + l * CCH, acc, NNODES, CCH, CCH);
    gemm_wmma<false><<<gE2, blk, 0, stream>>>(eb, we  + wo, (const float*)nullptr, eeb, NEDGES, CCH, CCH);

    // segment softmax attention
    attn_init   <<<(nh + 255) / 256, 256, 0, stream>>>(smax, den, nh);
    attn_score  <<<(eh + 255) / 256, 256, 0, stream>>>(qb, kb, eeb, src, dst, sc, smax);
    attn_exp    <<<(eh + 255) / 256, 256, 0, stream>>>(sc, smax, den, dst);
    attn_scatter<<<(ec + 255) / 256, 256, 0, stream>>>(sc, den, vb, eeb, src, dst, acc);

    add_layernorm<<<NNODES, CCH, 0, stream>>>(hb, acc, ln1g + l * CCH, ln1b + l * CCH);

    // FFN
    gemm_wmma<true ><<<gN8, blk, 0, stream>>>(hb, fw1 + (size_t)l * CCH * 512, fb1 + l * 512, ff,  NNODES, CCH, 512);
    gemm_wmma<false><<<gN2, blk, 0, stream>>>(ff, fw2 + (size_t)l * 512 * CCH, fb2 + l * CCH, acc, NNODES, 512, CCH);

    add_layernorm<<<NNODES, CCH, 0, stream>>>(hb, acc, ln2g + l * CCH, ln2b + l * CCH);
  }

  // edge MLP
  mlp1_wmma<<<dim3((NEDGES + 15) / 16, 1), blk, 0, stream>>>(hb, eb, src, dst, mw1, mb1, z1);
  mlp_tail <<<(NEDGES + 255) / 256, 256, 0, stream>>>(z1, mw2, mb2, mw3, mb3, out);
}
